// Net2_4638564680160
// MI455X (gfx1250) — compile-verified, matrix operand order checked by reading, not run on hardware
//
#include <hip/hip_runtime.h>
#include <hip/hip_bf16.h>

// ---------------------------------------------------------------------------
// Problem constants (from the reference):
//  B=512, IN=2000 (pad->2016), H1=1024, H2=128, O=64, F=8, FH=32, D=2112
// ---------------------------------------------------------------------------

typedef __bf16 bf16_t;
typedef __attribute__((ext_vector_type(8)))  __bf16 bf16x8;
typedef __attribute__((ext_vector_type(16))) __bf16 bf16x16;
typedef __attribute__((ext_vector_type(8)))  float  f32x8;

#define WMMA_BF16(a, b, c) \
  __builtin_amdgcn_wmma_f32_16x16x32_bf16(false, (a), false, (b), (short)0, (c), false, false)

// --------------------------- device helpers --------------------------------

__device__ __forceinline__ float eluf(float x) {
  return x > 0.f ? x : (__expf(x) - 1.f);
}
__device__ __forceinline__ float sigmf(float x) {
  return 1.f / (1.f + __expf(-x));
}

// A-fragment (16x32 bf16) from row-major [ld]; p points at (row0, k0).
// ISA layout: lanes 0-15 -> rows 0-15, K {0..7,16..23}; lanes 16-31 -> K {8..15,24..31}.
// Each half-fragment is 16 contiguous, 16B-aligned bytes -> two global_load_b128.
__device__ __forceinline__ bf16x16 ld_frag_a(const bf16_t* __restrict__ p, int ld, int lane) {
  const bf16_t* q = p + (lane & 15) * ld + ((lane & 16) ? 8 : 0);
  bf16x8 lo = *(const bf16x8*)(q);
  bf16x8 hi = *(const bf16x8*)(q + 16);
  return __builtin_shufflevector(lo, hi, 0, 1, 2, 3, 4, 5, 6, 7,
                                         8, 9, 10, 11, 12, 13, 14, 15);
}

// --------------------------- conversion ------------------------------------

__global__ void cvt_bf16(const float* __restrict__ src, bf16_t* __restrict__ dst,
                         int rows, int cols, int ldd) {
  int idx = blockIdx.x * blockDim.x + threadIdx.x;
  int total = rows * ldd;
  if (idx >= total) return;
  int r = idx / ldd, c = idx - r * ldd;
  dst[idx] = (c < cols) ? (bf16_t)src[r * cols + c] : (bf16_t)0.f;
}

// --------------------------- generic NT WMMA GEMM --------------------------
// C[M,N] = act(A[M,K] * B[N,K]^T + bias[N]); K multiple of 32, M,N of 16.
// block = 256 threads (8 waves); wave tile 16x16; block tile 32x64.
// HEADT: store transposed-per-head layout C[(m, (n&63)*64 + (n>>6))] (bf16 only),
//        used so the batched node-mix GEMM can consume B contiguously.

template <int ACT, bool OUT_BF16, bool HEADT>
__global__ void gemm_nt(const bf16_t* __restrict__ A, int lda,
                        const bf16_t* __restrict__ B, int ldb,
                        const float* __restrict__ bias,
                        void* __restrict__ C, int ldc,
                        int M, int N, int K) {
  const int lane = threadIdx.x & 31;
  const int wave = threadIdx.x >> 5;
  const int tn = blockIdx.x * 4 + (wave & 3);
  const int tm = blockIdx.y * 2 + (wave >> 2);
  const int m0 = tm * 16, n0 = tn * 16;
  if (m0 >= M || n0 >= N) return;

  const bf16_t* Ap = A + (size_t)m0 * lda;
  const bf16_t* Bp = B + (size_t)n0 * ldb;
  f32x8 acc = {};
  for (int k0 = 0; k0 < K; k0 += 32) {
    __builtin_prefetch(Ap + k0 + 128, 0, 1);   // global_prefetch_b8 on the K stream
    __builtin_prefetch(Bp + k0 + 128, 0, 1);
    bf16x16 af = ld_frag_a(Ap + k0, lda, lane);
    bf16x16 bf = ld_frag_a(Bp + k0, ldb, lane);
    acc = WMMA_BF16(af, bf, acc);
  }

  const int n = n0 + (lane & 15);
  const int mbase = m0 + ((lane & 16) ? 8 : 0);
  const float bs = bias ? bias[n] : 0.f;
  const int ncol = HEADT ? (((n & 63) << 6) + (n >> 6)) : n;
#pragma unroll
  for (int r = 0; r < 8; ++r) {
    float v = acc[r] + bs;
    if (ACT == 1) v = v > 0.f ? v : 0.f;
    if (OUT_BF16) ((bf16_t*)C)[(size_t)(mbase + r) * ldc + ncol] = (bf16_t)v;
    else          ((float*)C)[(size_t)(mbase + r) * ldc + ncol] = v;
  }
}

// --------------------------- attention chain (batch-independent) -----------

// cf[x,y,f,h] = elu(coef_m[x,y,f]*coef_nn_W[f,h] + coef_nn_b[f,h])   (1M elems)
__global__ void k_cf(const float* __restrict__ cm, const float* __restrict__ cw,
                     const float* __restrict__ cb, float* __restrict__ cf) {
  int idx = blockIdx.x * blockDim.x + threadIdx.x;
  if (idx >= 64 * 64 * 8 * 32) return;
  int fh = idx & 255;            // f*32+h
  int xyf = idx >> 5;            // (x*64+y)*8+f
  cf[idx] = eluf(cm[xyf] * cw[fh] + cb[fh]);
}

// s[x,y] = sum_{f,h} exp(elu(cf))
__global__ void k_s(const float* __restrict__ cf, float* __restrict__ s) {
  int idx = blockIdx.x * blockDim.x + threadIdx.x;
  if (idx >= 64 * 64) return;
  const float* p = cf + (size_t)idx * 256;
  float acc = 0.f;
  for (int i = 0; i < 256; ++i) acc += __expf(eluf(p[i]));
  s[idx] = acc;
}

// cfs[x,y,h] = elu( (1/F) * sum_f elu( exp(elu(cf[x,y,f,h])) * lin / s[x,y] ) ),
// lin = sum_h2 cf[x,y,f,h2]*coef_att_W[f,h,h2]
__global__ void k_cfs(const float* __restrict__ cf, const float* __restrict__ s,
                      const float* __restrict__ caW, float* __restrict__ cfs) {
  int idx = blockIdx.x * blockDim.x + threadIdx.x;
  if (idx >= 64 * 64 * 32) return;
  int h = idx & 31, xy = idx >> 5;
  float sv = s[xy], acc = 0.f;
  const float* cfb = cf + (size_t)xy * 256;
  for (int f = 0; f < 8; ++f) {
    const float* cfr = cfb + f * 32;
    const float* wr = caW + f * 1024 + h * 32;
    float lin = 0.f;
#pragma unroll
    for (int h2 = 0; h2 < 32; ++h2) lin += cfr[h2] * wr[h2];
    float e = __expf(eluf(cfr[h]));
    acc += eluf(e * lin / sv);
  }
  cfs[idx] = eluf(acc * 0.125f);
}

// coef_sum[o,h] = sum_i exp(elu(cfs[o,i,h]))
__global__ void k_csum(const float* __restrict__ cfs, float* __restrict__ cs) {
  int idx = blockIdx.x * blockDim.x + threadIdx.x;
  if (idx >= 64 * 32) return;
  int h = idx & 31, o = idx >> 5;
  float acc = 0.f;
  for (int i = 0; i < 64; ++i) acc += __expf(eluf(cfs[(o * 64 + i) * 32 + h]));
  cs[idx] = acc;
}

// att[o, i*32+g] = (exp(elu(cfs[o,i,g]))/coef_sum[o,g]) * sum_h cfs[o,i,h]*out_att_W[i,g,h]
__global__ void k_att(const float* __restrict__ cfs, const float* __restrict__ cs,
                      const float* __restrict__ oaW, float* __restrict__ att) {
  int idx = blockIdx.x * blockDim.x + threadIdx.x;
  if (idx >= 64 * 64 * 32) return;
  int g = idx & 31, oi = idx >> 5, i = oi & 63, o = oi >> 6;
  const float* cr = cfs + (size_t)oi * 32;
  const float* wr = oaW + i * 1024 + g * 32;
  float lin2 = 0.f;
#pragma unroll
  for (int h = 0; h < 32; ++h) lin2 += cr[h] * wr[h];
  float p2 = __expf(eluf(cr[g])) / cs[o * 32 + g];
  att[o * 2048 + i * 32 + g] = p2 * lin2;
}

// att2[o,j] = sum_p nodes_trans_W[o,p]*att[p,j]  -> bf16
__global__ void k_att2(const float* __restrict__ Wn, const float* __restrict__ att,
                       bf16_t* __restrict__ att2b) {
  int idx = blockIdx.x * blockDim.x + threadIdx.x;
  if (idx >= 64 * 2048) return;
  int j = idx & 2047, o = idx >> 11;
  float acc = 0.f;
  for (int p = 0; p < 64; ++p) acc += Wn[o * 64 + p] * att[p * 2048 + j];
  att2b[idx] = (bf16_t)acc;
}

// --------------------------- batched node mixing ---------------------------
// nt_h[b,o,j] = sum_p Wn[o,p]*hidT[b,j,p] : per-b 64x64x64 GEMM, both operands
// row-major over K (hidT produced pre-transposed by the heads GEMM).
// One block (16 waves / 512 thr) per batch element; wave -> one 16x16 tile.
__global__ void node_mix(const bf16_t* __restrict__ Wnb, const bf16_t* __restrict__ HdT,
                         bf16_t* __restrict__ NT) {
  const int lane = threadIdx.x & 31;
  const int wave = threadIdx.x >> 5;         // 0..15
  const int b = blockIdx.x;
  const int m0 = (wave >> 2) * 16, n0 = (wave & 3) * 16;
  const bf16_t* Hb = HdT + (size_t)b * 4096;
  f32x8 acc = {};
#pragma unroll
  for (int k0 = 0; k0 < 64; k0 += 32) {
    bf16x16 af = ld_frag_a(Wnb + m0 * 64 + k0, 64, lane);
    bf16x16 bf = ld_frag_a(Hb + n0 * 64 + k0, 64, lane);
    acc = WMMA_BF16(af, bf, acc);
  }
  bf16_t* out = NT + (size_t)b * 4096;
  const int n = n0 + (lane & 15);
  const int mbase = m0 + ((lane & 16) ? 8 : 0);
#pragma unroll
  for (int r = 0; r < 8; ++r) out[(mbase + r) * 64 + n] = (bf16_t)acc[r];
}

// --------------------------- fused final stage -----------------------------
// z[b*64+o] = prelu( ob + sum_e sigmoid( nt_h[row,:]·G[e,0:64] + c[o,e] ) * u[e] )
// A = nt_h [32768 x 64] bf16 ; B rows = Gb [2112 x 2112] bf16 (first 64 cols).
// Wave handles a 16-row tile, loops all 132 N-tiles, reduces in registers.
__global__ void final_fused(const bf16_t* __restrict__ NT, const bf16_t* __restrict__ Gb,
                            const float* __restrict__ Cmat, const float* __restrict__ u,
                            const float* __restrict__ ob, const float* __restrict__ pa,
                            float* __restrict__ z) {
  const int lane = threadIdx.x & 31;
  const int wave = threadIdx.x >> 5;
  const int m0 = (blockIdx.x * 8 + wave) * 16;     // 2048 tiles

  const bf16x16 a0 = ld_frag_a(NT + (size_t)m0 * 64, 64, lane);
  const bf16x16 a1 = ld_frag_a(NT + (size_t)m0 * 64 + 32, 64, lane);

  const int moff = (lane & 16) ? 8 : 0;
  const int ncol = lane & 15;
  float zacc[8] = {0.f, 0.f, 0.f, 0.f, 0.f, 0.f, 0.f, 0.f};

  for (int tn = 0; tn < 132; ++tn) {
    const int n0 = tn * 16;
    const bf16_t* Bp = Gb + (size_t)n0 * 2112;
    __builtin_prefetch(Bp + 16 * 2112, 0, 1);
    bf16x16 b0 = ld_frag_a(Bp, 2112, lane);
    bf16x16 b1 = ld_frag_a(Bp + 32, 2112, lane);
    f32x8 acc = {};
    acc = WMMA_BF16(a0, b0, acc);
    acc = WMMA_BF16(a1, b1, acc);

    const int e = n0 + ncol;
    const float ue = u[e];
#pragma unroll
    for (int r = 0; r < 8; ++r) {
      const int row = m0 + moff + r;
      const int o = row & 63;
      float t = acc[r] + Cmat[o * 2112 + e];
      zacc[r] += sigmf(t) * ue;
    }
  }

  // reduce the 16 columns held by the 16 lanes of each half-wave
#pragma unroll
  for (int r = 0; r < 8; ++r) {
    float v = zacc[r];
#pragma unroll
    for (int off = 1; off < 16; off <<= 1) v += __shfl_xor(v, off, 32);
    zacc[r] = v;
  }
  if ((lane & 15) == 0) {
    const float b0 = *ob, a = *pa;
#pragma unroll
    for (int r = 0; r < 8; ++r) {
      float v = zacc[r] + b0;
      z[m0 + moff + r] = v >= 0.f ? v : a * v;
    }
  }
}

// --------------------------- host launcher ---------------------------------

extern "C" void kernel_launch(void* const* d_in, const int* in_sizes, int n_in,
                              void* d_out, int out_size, void* d_ws, size_t ws_size,
                              hipStream_t stream) {
  const float* x   = (const float*)d_in[0];   // [512,2000]
  const float* Win = (const float*)d_in[1];   // [1024,2000]
  const float* bin = (const float*)d_in[2];   // [1024]
  const float* W1  = (const float*)d_in[3];   // [128,1024]
  const float* b1  = (const float*)d_in[4];   // [128]
  const float* Wo1 = (const float*)d_in[5];   // [64,64,128] -> [4096,128]
  const float* bo1 = (const float*)d_in[6];   // [4096]
  const float* cnW = (const float*)d_in[7];   // [8,32,1]
  const float* cnB = (const float*)d_in[8];   // [8,32]
  const float* caW = (const float*)d_in[9];   // [8,32,32]
  const float* oaW = (const float*)d_in[10];  // [64,32,32]
  const float* Wn  = (const float*)d_in[11];  // [64,64]
  const float* G   = (const float*)d_in[12];  // [2112,2112]
  const float* u   = (const float*)d_in[13];  // [2112]
  const float* ob  = (const float*)d_in[14];  // [1]
  const float* pa  = (const float*)d_in[15];  // [1]
  const float* cm  = (const float*)d_in[16];  // [64,64,8]
  float* z = (float*)d_out;                   // [512*64]

  char* w = (char*)d_ws;
  size_t off = 0;
  auto carve = [&](size_t bytes) -> char* {
    char* p = w + off;
    off += (bytes + 255) & ~(size_t)255;
    return p;
  };
  bf16_t* xb    = (bf16_t*)carve((size_t)512 * 2016 * 2);
  bf16_t* Winb  = (bf16_t*)carve((size_t)1024 * 2016 * 2);
  bf16_t* h1b   = (bf16_t*)carve((size_t)512 * 1024 * 2);
  bf16_t* W1b   = (bf16_t*)carve((size_t)128 * 1024 * 2);
  bf16_t* h2b   = (bf16_t*)carve((size_t)512 * 128 * 2);
  bf16_t* Wo1b  = (bf16_t*)carve((size_t)4096 * 128 * 2);
  bf16_t* hidT  = (bf16_t*)carve((size_t)512 * 4096 * 2);   // [b][j][p]
  bf16_t* Gb    = (bf16_t*)carve((size_t)2112 * 2112 * 2);
  bf16_t* Wnb   = (bf16_t*)carve((size_t)64 * 64 * 2);
  float*  cf    = (float*)carve((size_t)64 * 64 * 8 * 32 * 4);
  float*  sbuf  = (float*)carve((size_t)64 * 64 * 4);
  float*  cfs   = (float*)carve((size_t)64 * 64 * 32 * 4);
  float*  csum  = (float*)carve((size_t)64 * 32 * 4);
  float*  att   = (float*)carve((size_t)64 * 2048 * 4);
  bf16_t* att2b = (bf16_t*)carve((size_t)64 * 2048 * 2);
  float*  cmat  = (float*)carve((size_t)64 * 2112 * 4);
  bf16_t* nth   = (bf16_t*)carve((size_t)32768 * 64 * 2);
  (void)ws_size; (void)n_in; (void)in_sizes; (void)out_size;

  auto g1 = [](int n) { return (n + 255) / 256; };

  // fp32 -> bf16 (with K zero-padding where needed)
  cvt_bf16<<<g1(512 * 2016), 256, 0, stream>>>(x, xb, 512, 2000, 2016);
  cvt_bf16<<<g1(1024 * 2016), 256, 0, stream>>>(Win, Winb, 1024, 2000, 2016);
  cvt_bf16<<<g1(128 * 1024), 256, 0, stream>>>(W1, W1b, 128, 1024, 1024);
  cvt_bf16<<<g1(4096 * 128), 256, 0, stream>>>(Wo1, Wo1b, 4096, 128, 128);
  cvt_bf16<<<g1(2112 * 2112), 256, 0, stream>>>(G, Gb, 2112, 2112, 2112);
  cvt_bf16<<<g1(64 * 64), 256, 0, stream>>>(Wn, Wnb, 64, 64, 64);

  // trunk: h1 = relu(x @ Win^T + bin)   [512,1024]
  gemm_nt<1, true, false><<<dim3((1024 + 63) / 64, (512 + 31) / 32), 256, 0, stream>>>(
      xb, 2016, Winb, 2016, bin, h1b, 1024, 512, 1024, 2016);
  // h2 = relu(h1 @ W1^T + b1)           [512,128]
  gemm_nt<1, true, false><<<dim3((128 + 63) / 64, (512 + 31) / 32), 256, 0, stream>>>(
      h1b, 1024, W1b, 1024, b1, h2b, 128, 512, 128, 1024);
  // hiddens = relu(h2 @ Wo1^T + bo1) -> hidT[b][j][p]  (head-transposed store)
  gemm_nt<1, true, true><<<dim3((4096 + 63) / 64, (512 + 31) / 32), 256, 0, stream>>>(
      h2b, 128, Wo1b, 128, bo1, hidT, 4096, 512, 4096, 128);

  // batch-independent attention chain
  k_cf  <<<g1(64 * 64 * 8 * 32), 256, 0, stream>>>(cm, cnW, cnB, cf);
  k_s   <<<g1(64 * 64), 256, 0, stream>>>(cf, sbuf);
  k_cfs <<<g1(64 * 64 * 32), 256, 0, stream>>>(cf, sbuf, caW, cfs);
  k_csum<<<g1(64 * 32), 256, 0, stream>>>(cfs, csum);
  k_att <<<g1(64 * 64 * 32), 256, 0, stream>>>(cfs, csum, oaW, att);
  k_att2<<<g1(64 * 2048), 256, 0, stream>>>(Wn, att, att2b);

  // c[o,e] = att2[o,:] · G[e,64:]  (once; [64,2112] fp32)
  gemm_nt<0, false, false><<<dim3((2112 + 63) / 64, (64 + 31) / 32), 256, 0, stream>>>(
      att2b, 2048, Gb + 64, 2112, (const float*)nullptr, cmat, 2112, 64, 2112, 2048);

  // nt_h[b] = Wn @ hiddens[b]   (batched 64^3, bf16 out)
  node_mix<<<512, 512, 0, stream>>>(Wnb, hidT, nth);

  // fused: sigmoid(nt_h·G[:, :64]^T + c) · u  + ob, PReLU -> z[512,64]
  final_fused<<<256, 256, 0, stream>>>(nth, Gb, cmat, u, ob, pa, z);
}
